// MetaES_LSTM_18494129176763
// MI455X (gfx1250) — matrix operand order repaired and checked by reading, not compiled, not run on hardware
//
#include <hip/hip_runtime.h>
#include <hip/hip_bf16.h>
#include <stdint.h>

// Problem dims
#define T_LEN 2048
#define I_DIM 512
#define H_DIM 2048
#define G_DIM 8192   // 4*H
#define H2_DIM 64

// Persistent scan config
#define NB_SCAN 128
#define NT 256
#define UNITS_PER_BLOCK 16   // H_DIM / NB_SCAN
#define ROWS_PER_BLOCK 64    // UNITS_PER_BLOCK * 4 gates

typedef __bf16 bf16_t;
typedef bf16_t v16bf __attribute__((ext_vector_type(16)));
typedef float  v8f   __attribute__((ext_vector_type(8)));
typedef unsigned int u32x4 __attribute__((ext_vector_type(4)));
typedef int          i32x4 __attribute__((ext_vector_type(4)));
typedef int          i32x8 __attribute__((ext_vector_type(8)));

__device__ __forceinline__ unsigned short f32_to_bf16_rte(float f) {
  unsigned u = __float_as_uint(f);
  u += 0x7fffu + ((u >> 16) & 1u);
  return (unsigned short)(u >> 16);
}
__device__ __forceinline__ float bf16lo(unsigned u){ return __uint_as_float(u << 16); }
__device__ __forceinline__ float bf16hi(unsigned u){ return __uint_as_float(u & 0xffff0000u); }
__device__ __forceinline__ float sigmoidf_(float x){ return 1.0f/(1.0f + __expf(-x)); }

// ---------------------------------------------------------------------------
// TDM: DMA one 16-row x 2048-col bf16 slab (contiguous rows, stride 2048)
// from global into LDS. D# packing per CDNA5 ISA.
// ---------------------------------------------------------------------------
__device__ __forceinline__ void tdm_load_slab16x2048(const unsigned short* gsrc,
                                                     unsigned lds_byte_off) {
  const unsigned long long ga = (unsigned long long)(uintptr_t)gsrc;
  u32x4 g0;
  g0[0] = 1u;                                   // count=1, user mode
  g0[1] = lds_byte_off;                         // lds_addr
  g0[2] = (unsigned)(ga & 0xffffffffu);         // global_addr[31:0]
  g0[3] = (unsigned)((ga >> 32) & 0x1ffffffu)   // global_addr[56:32]
        | (2u << 30);                           // type=2 ("image")
  i32x8 g1;
  g1[0] = 0x00010000;          // workgroup_mask=0, data_size=1 (2 bytes)
  g1[1] = (int)(2048u << 16);  // tensor_dim0[15:0]=2048 @bits63:48
  g1[2] = (int)(16u << 16);    // tensor_dim0[31:16]=0, tensor_dim1[15:0]=16 @bits95:80
  g1[3] = (int)(2048u << 16);  // tensor_dim1[31:16]=0, tile_dim0=2048 @bits127:112
  g1[4] = 16;                  // tile_dim1=16, tile_dim2=0
  g1[5] = 2048;                // tensor_dim0_stride[31:0]=2048
  g1[6] = 0;                   // stride hi, tensor_dim1_stride lo
  g1[7] = 0;
  i32x4 gz4 = {0, 0, 0, 0};
#if __has_include(<hip/amd_detail/amd_gfx1250_TDM.h>)
  i32x8 gz8 = {0, 0, 0, 0, 0, 0, 0, 0};
  __builtin_amdgcn_tensor_load_to_lds(g0, g1, gz4, gz4, gz8, 0);   // clang-23 6-arg
#else
  __builtin_amdgcn_tensor_load_to_lds(g0, g1, gz4, gz4, 0);        // ROCm 7.2 5-arg
#endif
}

// ---------------------------------------------------------------------------
// Kernel 1: convert x, W_ih (and W_hh when TDM path active) to bf16; zero the
// grid-barrier counters.
// ---------------------------------------------------------------------------
__global__ void prep_kernel(const float* __restrict__ x, const float* __restrict__ W_ih,
                            const float* __restrict__ W_hh,
                            unsigned short* __restrict__ xb, unsigned short* __restrict__ wb,
                            unsigned short* __restrict__ whhb,
                            unsigned* __restrict__ bar) {
  const size_t stride = (size_t)gridDim.x * blockDim.x;
  const size_t nWih = (size_t)G_DIM * I_DIM;     // 4.2M
  const size_t nX   = (size_t)T_LEN * I_DIM;     // 1.0M
  const size_t nWhh = (size_t)G_DIM * H_DIM;     // 16.8M
  for (size_t i = (size_t)blockIdx.x * blockDim.x + threadIdx.x; i < nWhh; i += stride) {
    if (whhb) whhb[i] = f32_to_bf16_rte(W_hh[i]);
    if (i < nWih) wb[i] = f32_to_bf16_rte(W_ih[i]);
    if (i < nX)   xb[i] = f32_to_bf16_rte(x[i]);
  }
  if (blockIdx.x == 0 && threadIdx.x == 0) { bar[0] = 0u; bar[1] = 0u; }
}

// ---------------------------------------------------------------------------
// Kernel 2: xwb[T,4H] = x @ W_ih.T + (b_ih + b_hh), bf16 WMMA, f32 accumulate.
// One wave per 16x16 output tile; K=512 swept in 16x16x32 WMMA steps.
// ---------------------------------------------------------------------------
__global__ __launch_bounds__(NT) void gemm_xw_kernel(
    const unsigned short* __restrict__ xb, const unsigned short* __restrict__ wb,
    const float* __restrict__ b_ih, const float* __restrict__ b_hh,
    float* __restrict__ xwb) {
  const int wid    = blockIdx.x * (NT / 32) + (threadIdx.x >> 5);
  const int tile_m = wid >> 9;         // 512 tiles across G_DIM
  const int tile_n = wid & 511;
  const int lane = threadIdx.x & 31;
  const int l16  = lane & 15;
  const int lh   = lane >> 4;

  const int m = tile_m * 16 + l16;     // A row held by this lane (both halves)
  const int n = tile_n * 16 + l16;     // B column == W_ih row

  const unsigned short* arow = xb + (size_t)m * I_DIM;
  const unsigned short* brow = wb + (size_t)n * I_DIM;

  v8f acc = {};
  for (int k0 = 0; k0 < I_DIM; k0 += 32) {
    union { unsigned u[8]; v16bf v; } A, B;
#pragma unroll
    for (int v = 0; v < 8; ++v) {
      // A 16x32 bf16 layout: lanes 0-15 K = {0..7,16..23}; lanes 16-31 K = {8..15,24..31}
      const int ka = (v >> 2) * 16 + lh * 8 + (v & 3) * 2;
      A.u[v] = *(const unsigned*)(arow + k0 + ka);
      // B 32x16 bf16 layout: lanes 0-15 K = {0..7,16..23}; lanes 16-31 K = {8..15,24..31}
      const int kb = (v >> 2) * 16 + lh * 8 + (v & 3) * 2;
      B.u[v] = *(const unsigned*)(brow + k0 + kb);
    }
    acc = __builtin_amdgcn_wmma_f32_16x16x32_bf16(false, A.v, false, B.v,
                                                  (short)0, acc, false, false);
  }

  const float bias = b_ih[n] + b_hh[n];
#pragma unroll
  for (int r = 0; r < 8; ++r) {
    const int row = tile_m * 16 + r + lh * 8;   // C layout: VGPR r -> M = r (+8 upper half)
    xwb[(size_t)row * G_DIM + n] = acc[r] + bias;
  }
}

// ---------------------------------------------------------------------------
// Grid-wide barrier (generation counter in L2).
// ---------------------------------------------------------------------------
__device__ __forceinline__ void grid_barrier(unsigned* bar) {
  __syncthreads();
  __threadfence();
  if (threadIdx.x == 0) {
    const unsigned g = atomicAdd(&bar[1], 0u);
    const unsigned a = atomicAdd(&bar[0], 1u);
    if (a == NB_SCAN - 1) {
      atomicExch(&bar[0], 0u);
      atomicAdd(&bar[1], 1u);
    } else {
      while (atomicAdd(&bar[1], 0u) == g) __builtin_amdgcn_s_sleep(2);
    }
  }
  __syncthreads();
}

// ---------------------------------------------------------------------------
// Kernel 3: persistent LSTM scan. Each block owns 16 hidden units; its 64
// W_hh rows (bf16, 256 KiB, gate-major: row = gate*16 + unit) live in LDS for
// all 4096 steps, staged by the Tensor Data Mover (4 slabs of 16x2048 bf16)
// when a bf16 W_hh copy is available, else converted in-kernel. Double-
// buffered global h, block-local c, grid barrier per step. Both reference
// passes collapse into one 4096-step loop (xw row = t mod 2048).
// ---------------------------------------------------------------------------
__global__ __launch_bounds__(NT) void lstm_scan_kernel(
    const float* __restrict__ h0, const float* __restrict__ c0,
    const float* __restrict__ W_hh, const unsigned short* __restrict__ whhb,
    const float* __restrict__ xwb,
    const float* __restrict__ W1, const float* __restrict__ b1,
    const float* __restrict__ W2, const float* __restrict__ b2,
    float* __restrict__ hbuf0, float* __restrict__ hbuf1,
    unsigned* __restrict__ bar, float* __restrict__ out, int use_tdm) {
  __shared__ __align__(16) unsigned short w_sh[ROWS_PER_BLOCK * H_DIM]; // 256 KiB
  __shared__ __align__(16) float h_sh[H_DIM];                           // 8 KiB
  __shared__ float c_sh[UNITS_PER_BLOCK];
  __shared__ float a_sh[H2_DIM];

  const int tid  = threadIdx.x;
  const int wid  = tid >> 5;
  const int lane = tid & 31;
  const int blk  = blockIdx.x;

  if (use_tdm) {
    // Async-tensor staging: 4 contiguous 16-row slabs (one per gate).
    if (wid == 0) {
      const unsigned w_base = (unsigned)(uintptr_t)(&w_sh[0]);
#pragma unroll
      for (int g = 0; g < 4; ++g) {
        const unsigned short* src =
            whhb + ((size_t)g * H_DIM + (size_t)blk * UNITS_PER_BLOCK) * H_DIM;
        tdm_load_slab16x2048(src, w_base + (unsigned)(g * UNITS_PER_BLOCK * H_DIM * 2));
      }
      __builtin_amdgcn_s_wait_tensorcnt(0);
    }
  } else {
    // Fallback: convert f32 W_hh into LDS directly (same gate-major layout).
    for (int idx = tid; idx < ROWS_PER_BLOCK * H_DIM; idx += NT) {
      const int r = idx >> 11;
      const int col = idx & (H_DIM - 1);
      const int gate = r >> 4, unit_local = r & 15;
      const int grow = gate * H_DIM + (blk * UNITS_PER_BLOCK + unit_local);
      w_sh[idx] = f32_to_bf16_rte(W_hh[(size_t)grow * H_DIM + col]);
    }
  }
  if (tid < UNITS_PER_BLOCK) c_sh[tid] = c0[blk * UNITS_PER_BLOCK + tid];
  if (blk == 0) {
    for (int i = tid; i < H_DIM; i += NT) hbuf0[i] = h0[i];
  }
  grid_barrier(bar);

  float* hb[2] = { hbuf0, hbuf1 };
  const uint4* wsh4 = reinterpret_cast<const uint4*>(w_sh);

  for (int t = 0; t < 2 * T_LEN; ++t) {
    const float* hin = hb[t & 1];
    float*      hout = hb[(t + 1) & 1];
    for (int i = tid; i < H_DIM; i += NT) h_sh[i] = hin[i];
    __syncthreads();

    // Each wave handles 2 units; acc[rr]: gate = rr&3, unit = wid*2 + (rr>>2),
    // LDS row (gate-major) = (rr&3)*16 + wid*2 + (rr>>2).
    float acc[8];
#pragma unroll
    for (int r = 0; r < 8; ++r) acc[r] = 0.0f;

#pragma unroll 2
    for (int it = 0; it < 8; ++it) {
      const int chunk = it * 32 + lane;        // uint4 index within a row (0..255)
      const int k = chunk << 3;                // element base (8 bf16 per uint4)
      const float4 ha  = *reinterpret_cast<const float4*>(&h_sh[k]);
      const float4 hbv = *reinterpret_cast<const float4*>(&h_sh[k + 4]);
#pragma unroll
      for (int rr = 0; rr < 8; ++rr) {
        const int row = (rr & 3) * 16 + wid * 2 + (rr >> 2);
        const uint4 w = wsh4[(size_t)row * 256 + chunk];
        float s = acc[rr];
        s = fmaf(bf16lo(w.x), ha.x,  s);
        s = fmaf(bf16hi(w.x), ha.y,  s);
        s = fmaf(bf16lo(w.y), ha.z,  s);
        s = fmaf(bf16hi(w.y), ha.w,  s);
        s = fmaf(bf16lo(w.z), hbv.x, s);
        s = fmaf(bf16hi(w.z), hbv.y, s);
        s = fmaf(bf16lo(w.w), hbv.z, s);
        s = fmaf(bf16hi(w.w), hbv.w, s);
        acc[rr] = s;
      }
    }

    // wave32 butterfly reduction
#pragma unroll
    for (int r = 0; r < 8; ++r) {
#pragma unroll
      for (int off = 16; off > 0; off >>= 1)
        acc[r] += __shfl_xor(acc[r], off, 32);
    }

    if (lane < 2) {
      const int ul = wid * 2 + lane;                  // local unit 0..15
      const int j  = blk * UNITS_PER_BLOCK + ul;      // global unit
      const float* xrow = xwb + (size_t)(t & (T_LEN - 1)) * G_DIM;
      const float gi = sigmoidf_(xrow[0 * H_DIM + j] + acc[lane * 4 + 0]);
      const float gf = sigmoidf_(xrow[1 * H_DIM + j] + acc[lane * 4 + 1]);
      const float gg = tanhf   (xrow[2 * H_DIM + j] + acc[lane * 4 + 2]);
      const float go = sigmoidf_(xrow[3 * H_DIM + j] + acc[lane * 4 + 3]);
      const float c  = gf * c_sh[ul] + gi * gg;
      c_sh[ul] = c;
      hout[j] = go * tanhf(c);
      // Prefetch next step's gate preactivations into cache (global_prefetch_b8).
      const float* xnext = xwb + (size_t)((t + 1) & (T_LEN - 1)) * G_DIM;
#pragma unroll
      for (int g = 0; g < 4; ++g) __builtin_prefetch(xnext + g * H_DIM + j, 0, 3);
    }
    grid_barrier(bar);
  }

  // Policy head: sigmoid((h2 @ W1.T + b1) @ W2.T + b2) * 0.06 + 0.02
  if (blk == 0) {
    const float* h2 = hb[0];  // final h after 4096 steps lands back in hbuf0
    if (tid < H2_DIM) {
      float s = b1[tid];
      const float* wr = W1 + (size_t)tid * H_DIM;
      for (int k = 0; k < H_DIM; ++k) s = fmaf(h2[k], wr[k], s);
      a_sh[tid] = s;
    }
    __syncthreads();
    if (tid == 0) {
      float z = b2[0];
      for (int jj = 0; jj < H2_DIM; ++jj) z = fmaf(a_sh[jj], W2[jj], z);
      out[0] = sigmoidf_(z) * 0.06f + 0.02f;
    }
  }
}

// ---------------------------------------------------------------------------
extern "C" void kernel_launch(void* const* d_in, const int* in_sizes, int n_in,
                              void* d_out, int out_size, void* d_ws, size_t ws_size,
                              hipStream_t stream) {
  (void)in_sizes; (void)n_in; (void)out_size;
  const float* x    = (const float*)d_in[0];
  const float* h0   = (const float*)d_in[1];
  const float* c0   = (const float*)d_in[2];
  const float* W_ih = (const float*)d_in[3];
  const float* W_hh = (const float*)d_in[4];
  const float* b_ih = (const float*)d_in[5];
  const float* b_hh = (const float*)d_in[6];
  const float* W1   = (const float*)d_in[7];
  const float* b1   = (const float*)d_in[8];
  const float* W2   = (const float*)d_in[9];
  const float* b2   = (const float*)d_in[10];
  float* out = (float*)d_out;
  char*  ws  = (char*)d_ws;

  // Workspace layout (TDM path needs an extra 32 MiB for bf16 W_hh).
  const size_t XB_SZ   = (size_t)T_LEN * I_DIM * 2;   // 2 MiB
  const size_t WIHB_SZ = (size_t)G_DIM * I_DIM * 2;   // 8 MiB
  const size_t WHHB_SZ = (size_t)G_DIM * H_DIM * 2;   // 32 MiB
  const size_t XWB_SZ  = (size_t)T_LEN * G_DIM * 4;   // 64 MiB
  const size_t need_tdm = XB_SZ + WIHB_SZ + WHHB_SZ + XWB_SZ + 2 * 8192 + 256;
  const int use_tdm = (ws_size >= need_tdm) ? 1 : 0;

  size_t off = 0;
  unsigned short* xb    = (unsigned short*)(ws + off); off += XB_SZ;
  unsigned short* wihb  = (unsigned short*)(ws + off); off += WIHB_SZ;
  unsigned short* whhb  = nullptr;
  if (use_tdm) { whhb = (unsigned short*)(ws + off); off += WHHB_SZ; }
  float*    xwb   = (float*)(ws + off); off += XWB_SZ;
  float*    hbuf0 = (float*)(ws + off); off += 8192;
  float*    hbuf1 = (float*)(ws + off); off += 8192;
  unsigned* bar   = (unsigned*)(ws + off);

  prep_kernel<<<4096, NT, 0, stream>>>(x, W_ih, W_hh, xb, wihb, whhb, bar);

  // 128 x 512 = 65536 16x16 tiles, 8 waves (one tile each) per 256-thread block
  gemm_xw_kernel<<<(T_LEN / 16) * (G_DIM / 16) / (NT / 32), NT, 0, stream>>>(
      xb, wihb, b_ih, b_hh, xwb);

  lstm_scan_kernel<<<NB_SCAN, NT, 0, stream>>>(
      h0, c0, W_hh, whhb, xwb, W1, b1, W2, b2, hbuf0, hbuf1, bar, out, use_tdm);
}
